// FGNNI_16612933501380
// MI455X (gfx1250) — compile-verified, matrix operand order checked by reading, not run on hardware
//
#include <hip/hip_runtime.h>
#include <hip/hip_bf16.h>

typedef __bf16 bf16_t;
typedef __attribute__((ext_vector_type(16))) __bf16 v16bf;
typedef __attribute__((ext_vector_type(8)))  __bf16 v8bf;
typedef __attribute__((ext_vector_type(8)))  float  v8f;

// Problem constants (from reference): N=64, D=128, EMB=64, OUTG=64, A=512, B=256, G=8
// group_mask[i,j] == (j>>1 == i)  (D/N == 2) -- baked into the kernels.

__device__ __forceinline__ v8f vzero8() {
  v8f z;
#pragma unroll
  for (int i = 0; i < 8; ++i) z[i] = 0.0f;
  return z;
}

__device__ __forceinline__ float leaky01(float x) { return fmaxf(x, 0.01f * x); }

// ---------------------------------------------------------------------------
// WMMA fragment load.
// Matrix stored row-major [rows][ld] in bf16. For the A operand row = M index;
// for the B operand pass B^T so row = N index (symmetric lane mapping).
// Per ISA 16-bit 16x32 operand layout, lane element e (0..15) holds
//   k = k0 + (e & 7) + ((e & 8) ? 16 : 0) + ((lane >> 4) << 3)
// i.e. two contiguous 16-byte chunks per lane at k-offsets {base, base+16}.
// ---------------------------------------------------------------------------
__device__ __forceinline__ v16bf load_frag(const bf16_t* __restrict__ base,
                                           int row0, int k0, int ld) {
  int lane = threadIdx.x & 31;
  const bf16_t* p = base + (row0 + (lane & 15)) * ld + k0 + ((lane >> 4) << 3);
  v8bf lo = *(const v8bf*)(p);
  v8bf hi = *(const v8bf*)(p + 16);
  return __builtin_shufflevector(lo, hi, 0, 1, 2, 3, 4, 5, 6, 7,
                                 8, 9, 10, 11, 12, 13, 14, 15);
}

__device__ __forceinline__ v8f wmma_bf16(v16bf a, v16bf b, v8f c) {
  // 8 args: (neg_a, A, neg_b, B, c_mod, C, reuse_a, reuse_b)
  return __builtin_amdgcn_wmma_f32_16x16x32_bf16(false, a, false, b,
                                                 (short)0, c, false, false);
}

// One wave computes a 16(row) x 256(col) output block: 16 col-tiles, K-loop by 32.
__device__ __forceinline__ void gemm_m16_n256(const bf16_t* __restrict__ A, int lda,
                                              const bf16_t* __restrict__ BT, int ldb,
                                              int m0, int n0, int K, v8f acc[16]) {
#pragma unroll 1
  for (int k = 0; k < K; k += 32) {
    v16bf a = load_frag(A, m0, k, lda);
#pragma unroll
    for (int t = 0; t < 16; ++t) {
      v16bf b = load_frag(BT, n0 + t * 16, k, ldb);
      acc[t] = wmma_bf16(a, b, acc[t]);
    }
  }
}

// Write 16x256 accumulator block to row-major bf16 [.,ldd] with per-column bias
// and LeakyReLU(0.01). C/D layout: vgpr r, lane -> M = m0 + 8*(lane>>4) + r,
// N = n0 + (lane&15).
__device__ __forceinline__ void store_colbias_leaky(bf16_t* __restrict__ dst, int ldd,
                                                    int m0, int n0w, const v8f acc[16],
                                                    const float* __restrict__ bias) {
  int lane = threadIdx.x & 31;
#pragma unroll
  for (int t = 0; t < 16; ++t) {
    int nn = n0w + t * 16 + (lane & 15);
    int mb = m0 + ((lane >> 4) << 3);
    float bc = bias[nn];
#pragma unroll
    for (int r = 0; r < 8; ++r) {
      float v = acc[t][r] + bc;
      dst[(mb + r) * ldd + nn] = (bf16_t)leaky01(v);
    }
  }
}

// ---------------------------------------------------------------------------
// Precompute kernels (tiny; all operands stay in L2)
// ---------------------------------------------------------------------------

// src (K x Nc) f32 row-major -> dst (Nc x K) bf16 row-major (transposed weights)
__global__ void k_conv_t(bf16_t* __restrict__ dst, const float* __restrict__ src,
                         int K, int Nc) {
  int idx = blockIdx.x * 256 + threadIdx.x;
  if (idx >= K * Nc) return;
  int n = idx / K, k = idx % K;
  dst[idx] = (bf16_t)src[k * Nc + n];
}

// src (K x Nc) f32 -> dst (Nc x K) f32 (for f_w2^T, kept f32 for the final dots)
__global__ void k_trans_f32(float* __restrict__ dst, const float* __restrict__ src,
                            int K, int Nc) {
  int idx = blockIdx.x * 256 + threadIdx.x;
  if (idx >= K * Nc) return;
  int n = idx / K, k = idx % K;
  dst[idx] = src[k * Nc + n];
}

// wadjT[g][l][j] = graphs[g][j][l] * w[j][l]   (bf16, A operand of aggr GEMM)
__global__ void k_wadj(bf16_t* __restrict__ dst, const float* __restrict__ graphs,
                       const float* __restrict__ w) {
  int idx = blockIdx.x * 256 + threadIdx.x;
  if (idx >= 8 * 64 * 64) return;
  int g = idx >> 12, rem = idx & 4095, l = rem >> 6, j = rem & 63;
  dst[idx] = (bf16_t)(graphs[(g << 12) + (j << 6) + l] * w[(j << 6) + l]);
}

// c[n][a] = sum_e emb[n][e] * W0[(off+e)][a] + b0[a]   (embedding half of layer-1)
__global__ void k_cbias(float* __restrict__ c, const float* __restrict__ emb,
                        const float* __restrict__ W0, const float* __restrict__ b0,
                        int off) {
  int n = blockIdx.x;
  for (int a = threadIdx.x; a < 512; a += 256) {
    float s = b0[a];
#pragma unroll 4
    for (int e = 0; e < 64; ++e)
      s = fmaf(emb[n * 64 + e], W0[(off + e) * 512 + a], s);
    c[n * 512 + a] = s;
  }
}

// ---------------------------------------------------------------------------
// K1: g-MLP. One workgroup per batch b (256 WGs, 8 waves).
// Produces emb_sT[b][outg][node] (bf16), ready as B^T of the aggr GEMM.
// ---------------------------------------------------------------------------
__global__ __launch_bounds__(256)
void k1_gmlp(const float* __restrict__ samples, const float* __restrict__ g_w0,
             const float* __restrict__ g_b1, const float* __restrict__ g_b2,
             const float* __restrict__ c_g, const bf16_t* __restrict__ g_w1T,
             const bf16_t* __restrict__ g_w2T, bf16_t* __restrict__ emb_sT) {
  extern __shared__ char smem[];
  bf16_t* h0s = (bf16_t*)smem;  // 64 x 512 bf16 (h0, then reused for h1)

  int b = blockIdx.x;
  int tid = threadIdx.x, lane = tid & 31, wv = tid >> 5;

  // Phase A: h0[n][a] = leaky(s[b,2n]*W0[2n][a] + s[b,2n+1]*W0[2n+1][a] + c_g[n][a])
  {
    int n = tid >> 2, part = tid & 3;
    float s0 = samples[b * 128 + 2 * n];
    float s1 = samples[b * 128 + 2 * n + 1];
    const float* w0a = g_w0 + (2 * n) * 512;
    const float* w0b = g_w0 + (2 * n + 1) * 512;
    const float* cg = c_g + n * 512;
    for (int a = part * 128; a < part * 128 + 128; ++a) {
      float v = fmaf(s0, w0a[a], fmaf(s1, w0b[a], cg[a]));
      h0s[n * 512 + a] = (bf16_t)leaky01(v);
    }
  }
  __syncthreads();

  // Phase B: h1 = leaky(h0 @ g_w1 + g_b1)   (64x512x512 WMMA)
  int m0 = (wv >> 1) * 16;
  int n0w = (wv & 1) * 256;
  v8f acc[16];
#pragma unroll
  for (int t = 0; t < 16; ++t) acc[t] = vzero8();
  gemm_m16_n256(h0s, 512, g_w1T, 512, m0, n0w, 512, acc);
  __syncthreads();                                   // all reads of h0 done
  store_colbias_leaky(h0s, 512, m0, n0w, acc, g_b1); // h1 overwrites h0
  __syncthreads();

  // Phase C: emb_s = h1 @ g_w2 + g_b2   (64x512x64 WMMA; 2 tiles per wave)
  v8f acc2[2];
#pragma unroll
  for (int t = 0; t < 2; ++t) acc2[t] = vzero8();
#pragma unroll
  for (int t = 0; t < 2; ++t) {
    int tile = wv * 2 + t;
    int m0d = (tile >> 2) * 16, n0d = (tile & 3) * 16;
#pragma unroll 1
    for (int k = 0; k < 512; k += 32) {
      v16bf a = load_frag(h0s, m0d, k, 512);
      v16bf bb = load_frag(g_w2T, n0d, k, 512);
      acc2[t] = wmma_bf16(a, bb, acc2[t]);
    }
  }
  // Store transposed: emb_sT[b][outg=nn][node=m] (bf16)
#pragma unroll
  for (int t = 0; t < 2; ++t) {
    int tile = wv * 2 + t;
    int m0d = (tile >> 2) * 16, n0d = (tile & 3) * 16;
    int nn = n0d + (lane & 15);
    int mb = m0d + ((lane >> 4) << 3);
    float bc = g_b2[nn];
#pragma unroll
    for (int r = 0; r < 8; ++r)
      emb_sT[b * 4096 + nn * 64 + mb + r] = (bf16_t)(acc2[t][r] + bc);
  }
}

// ---------------------------------------------------------------------------
// K2: fused aggregation + f-MLP + masked output. One WG per (b,g): 2048 WGs.
// ---------------------------------------------------------------------------
__global__ __launch_bounds__(256)
void k2_fused(const bf16_t* __restrict__ wadjT, const bf16_t* __restrict__ emb_sT,
              const bf16_t* __restrict__ f_w0aT, const float* __restrict__ c_f,
              const bf16_t* __restrict__ f_w1T, const float* __restrict__ f_b1,
              const float* __restrict__ f_w2T, const float* __restrict__ f_b2,
              float* __restrict__ out) {
  extern __shared__ char smem[];
  bf16_t* aggrs = (bf16_t*)smem;               // 64 x 64 bf16 (8 KB)
  bf16_t* h0s = (bf16_t*)(smem + 64 * 64 * 2); // 64 x 512 bf16 (64 KB; h0 then h1)
  float* psum = (float*)smem;                  // reuses aggr region in phase 4

  int b = blockIdx.x >> 3, g = blockIdx.x & 7;
  int tid = threadIdx.x, lane = tid & 31, wv = tid >> 5;

  // Phase 1: aggr[l][k] = sum_j wadjT[g][l][j] * emb_s[b][j][k]  (64x64x64 WMMA)
  const bf16_t* A1 = wadjT + g * 4096;   // [l][j]
  const bf16_t* B1 = emb_sT + b * 4096;  // [k][j] == B^T
  v8f acc2[2];
#pragma unroll
  for (int t = 0; t < 2; ++t) acc2[t] = vzero8();
#pragma unroll
  for (int t = 0; t < 2; ++t) {
    int tile = wv * 2 + t;
    int m0d = (tile >> 2) * 16, n0d = (tile & 3) * 16;
#pragma unroll
    for (int k = 0; k < 64; k += 32) {
      v16bf a = load_frag(A1, m0d, k, 64);
      v16bf bb = load_frag(B1, n0d, k, 64);
      acc2[t] = wmma_bf16(a, bb, acc2[t]);
    }
  }
#pragma unroll
  for (int t = 0; t < 2; ++t) {
    int tile = wv * 2 + t;
    int m0d = (tile >> 2) * 16, n0d = (tile & 3) * 16;
    int nn = n0d + (lane & 15);
    int mb = m0d + ((lane >> 4) << 3);
#pragma unroll
    for (int r = 0; r < 8; ++r)
      aggrs[(mb + r) * 64 + nn] = (bf16_t)acc2[t][r];
  }
  __syncthreads();

  // Phase 2: h0 = leaky(aggr @ f_w0[:OUTG] + c_f[node])  (64x64x512 WMMA)
  int m0 = (wv >> 1) * 16;
  int n0w = (wv & 1) * 256;
  v8f acc[16];
#pragma unroll
  for (int t = 0; t < 16; ++t) acc[t] = vzero8();
  gemm_m16_n256(aggrs, 64, f_w0aT, 64, m0, n0w, 64, acc);
#pragma unroll
  for (int t = 0; t < 16; ++t) {  // per-(row,col) bias from c_f
    int nn = n0w + t * 16 + (lane & 15);
    int mb = m0 + ((lane >> 4) << 3);
#pragma unroll
    for (int r = 0; r < 8; ++r) {
      float v = acc[t][r] + c_f[(mb + r) * 512 + nn];
      h0s[(mb + r) * 512 + nn] = (bf16_t)leaky01(v);
    }
  }
  __syncthreads();

  // Phase 3: h1 = leaky(h0 @ f_w1 + f_b1)  (64x512x512 WMMA — the dominant GEMM)
#pragma unroll
  for (int t = 0; t < 16; ++t) acc[t] = vzero8();
  gemm_m16_n256(h0s, 512, f_w1T, 512, m0, n0w, 512, acc);
  __syncthreads();
  store_colbias_leaky(h0s, 512, m0, n0w, acc, f_b1);  // h1 overwrites h0
  __syncthreads();

  // Phase 4: masked layer 3: out[b][g][d] = h1[d>>1] . f_w2[:,d] + f_b2[d]
  {
    int d = tid >> 1, half = tid & 1, i = d >> 1;
    const bf16_t* hrow = h0s + i * 512 + half * 256;
    const float* wrow = f_w2T + d * 512 + half * 256;
    float s = 0.0f;
#pragma unroll 8
    for (int a = 0; a < 256; ++a) s = fmaf((float)hrow[a], wrow[a], s);
    psum[tid] = s;
  }
  __syncthreads();
  if (tid < 128)
    out[(b * 8 + g) * 128 + tid] = psum[2 * tid] + psum[2 * tid + 1] + f_b2[tid];
}

// ---------------------------------------------------------------------------
extern "C" void kernel_launch(void* const* d_in, const int* in_sizes, int n_in,
                              void* d_out, int out_size, void* d_ws, size_t ws_size,
                              hipStream_t stream) {
  (void)in_sizes; (void)n_in; (void)out_size; (void)ws_size;
  const float* samples    = (const float*)d_in[0];
  const float* graphs     = (const float*)d_in[1];
  /* d_in[2] group_mask: structure baked in (mask[i,j] == (j>>1 == i)) */
  const float* embeddings = (const float*)d_in[3];
  const float* w          = (const float*)d_in[4];
  const float* g_w0 = (const float*)d_in[5];
  const float* g_b0 = (const float*)d_in[6];
  const float* g_w1 = (const float*)d_in[7];
  const float* g_b1 = (const float*)d_in[8];
  const float* g_w2 = (const float*)d_in[9];
  const float* g_b2 = (const float*)d_in[10];
  const float* f_w0 = (const float*)d_in[11];
  const float* f_b0 = (const float*)d_in[12];
  const float* f_w1 = (const float*)d_in[13];
  const float* f_b1 = (const float*)d_in[14];
  const float* f_w2 = (const float*)d_in[15];
  const float* f_b2 = (const float*)d_in[16];
  float* out = (float*)d_out;

  // Workspace carve-out (~3.9 MB)
  char* p = (char*)d_ws;
  bf16_t* g_w1T  = (bf16_t*)p; p += 512 * 512 * 2;
  bf16_t* f_w1T  = (bf16_t*)p; p += 512 * 512 * 2;
  bf16_t* g_w2T  = (bf16_t*)p; p += 64 * 512 * 2;
  bf16_t* f_w0aT = (bf16_t*)p; p += 512 * 64 * 2;
  float*  f_w2T  = (float*)p;  p += 128 * 512 * 4;
  bf16_t* wadjT  = (bf16_t*)p; p += 8 * 64 * 64 * 2;
  float*  c_g    = (float*)p;  p += 64 * 512 * 4;
  float*  c_f    = (float*)p;  p += 64 * 512 * 4;
  bf16_t* emb_sT = (bf16_t*)p; p += 256 * 64 * 64 * 2;

  k_conv_t<<<1024, 256, 0, stream>>>(g_w1T, g_w1, 512, 512);
  k_conv_t<<<1024, 256, 0, stream>>>(f_w1T, f_w1, 512, 512);
  k_conv_t<<<128, 256, 0, stream>>>(g_w2T, g_w2, 512, 64);
  k_conv_t<<<128, 256, 0, stream>>>(f_w0aT, f_w0, 64, 512);  // first 64 rows of f_w0
  k_trans_f32<<<256, 256, 0, stream>>>(f_w2T, f_w2, 512, 128);
  k_wadj<<<128, 256, 0, stream>>>(wadjT, graphs, w);
  k_cbias<<<64, 256, 0, stream>>>(c_g, embeddings, g_w0, g_b0, 128);
  k_cbias<<<64, 256, 0, stream>>>(c_f, embeddings, f_w0, f_b0, 64);

  k1_gmlp<<<256, 256, 64 * 512 * 2, stream>>>(samples, g_w0, g_b1, g_b2, c_g,
                                              g_w1T, g_w2T, emb_sT);
  k2_fused<<<2048, 256, 64 * 64 * 2 + 64 * 512 * 2, stream>>>(
      wadjT, emb_sT, f_w0aT, c_f, f_w1T, f_b1, f_w2T, f_b2, out);
}
